// GraphAttTrans_43757126811988
// MI455X (gfx1250) — compile-verified, matrix operand order checked by reading, not run on hardware
//
#include <hip/hip_runtime.h>

// ---- problem sizes ----
#define NN    496000          // total GAT nodes
#define EE    2000000         // real edges
#define ETOT  2496000         // edges + self loops
#define ROWS  1984            // B * NUM_NODES
#define TT    250
#define SEQ   17856           // ROWS * 9 (L after 3x conv stride-3)
#define DM    64
#define FFD   2048

typedef __attribute__((ext_vector_type(16))) _Float16 v16h;
typedef __attribute__((ext_vector_type(8)))  float    v8f;

// K index mapping for 16-bit A/B fragments of v_wmma_f32_16x16x32_f16
// lanes 0-15: K in {0..7, 16..23}; lanes 16-31: K in {8..15, 24..31}
__device__ __forceinline__ int kmap16(int j, int hf){ return (j < 8 ? j : j + 8) + (hf ? 8 : 0); }

__device__ __forceinline__ v8f wmma_f16(v16h a, v16h b, v8f c){
  return __builtin_amdgcn_wmma_f32_16x16x32_f16(false, a, false, b, (short)0, c, false, false);
}

// CDNA5 async global->LDS copy (tracked by ASYNCcnt, no VGPR data path).
// Per ISA 10.2, the low 32 bits of a generic pointer to __shared__ are the LDS byte address.
__device__ __forceinline__ void async_b128(unsigned lds, const float* g){
  asm volatile("global_load_async_to_lds_b128 %0, %1, off" :: "v"(lds), "v"(g) : "memory");
}
__device__ __forceinline__ void async_wait0(){
  asm volatile("s_wait_asynccnt 0x0" ::: "memory");
}

// monotonic float<->uint encoding for atomicMax on floats
__device__ __forceinline__ unsigned fenc(float f){
  unsigned u = __float_as_uint(f);
  return (u & 0x80000000u) ? ~u : (u | 0x80000000u);
}
__device__ __forceinline__ float fdec(unsigned u){
  return (u & 0x80000000u) ? __uint_as_float(u & 0x7FFFFFFFu) : __uint_as_float(~u);
}

// -------------------- utility --------------------
__global__ void k_zero(float* p, int n){
  int i = blockIdx.x * blockDim.x + threadIdx.x;
  if (i < n) p[i] = 0.f;
}

// -------------------- GAT: self-loop attr mean --------------------
__global__ void k_loop_attr(const int* __restrict__ ei, const float* __restrict__ attr,
                            float* loops, float* cnt){
  long long idx = (long long)blockIdx.x * blockDim.x + threadIdx.x;
  if (idx >= (long long)EE * 8) return;
  int e = (int)(idx >> 3), d = (int)(idx & 7);
  int dn = ei[EE + e];
  atomicAdd(&loops[(size_t)dn * 8 + d], attr[(size_t)e * 8 + d]);
  if (d == 0) atomicAdd(&cnt[dn], 1.f);
}

// w_e[d] = sum_c W_edge[d][c] * att_edge[c]   (since (ea@We)·ae == ea·(We@ae))
__global__ void k_edge_vec(const float* __restrict__ We, const float* __restrict__ ae,
                           float* wev, int C){
  int d = threadIdx.x;
  if (d < 8){
    float s = 0.f;
    for (int c = 0; c < C; ++c) s += We[d * C + c] * ae[c];
    wev[d] = s;
  }
}

__global__ void k_aeloop(const float* __restrict__ loops, const float* __restrict__ cnt,
                         const float* __restrict__ wev, float* out){
  int n = blockIdx.x * blockDim.x + threadIdx.x;
  if (n >= NN) return;
  float inv = 1.f / fmaxf(cnt[n], 1.f);
  float s = 0.f;
  for (int d = 0; d < 8; ++d) s += loops[(size_t)n * 8 + d] * inv * wev[d];
  out[n] = s;
}

// xs = h @ W ; a_src = xs·att_src ; a_dst = xs·att_dst
__global__ void k_xs(const float* __restrict__ h, const float* __restrict__ W,
                     const float* __restrict__ atts, const float* __restrict__ attd,
                     float* xs, float* a_src, float* a_dst, int Cin, int C){
  int n = blockIdx.x * blockDim.x + threadIdx.x;
  if (n >= NN) return;
  float hv[32];
  for (int k = 0; k < Cin; ++k) hv[k] = h[(size_t)n * Cin + k];
  float as = 0.f, ad = 0.f;
  for (int c = 0; c < C; ++c){
    float s = 0.f;
    for (int k = 0; k < Cin; ++k) s += hv[k] * W[k * C + c];
    xs[(size_t)n * C + c] = s;
    as += s * atts[c]; ad += s * attd[c];
  }
  a_src[n] = as; a_dst[n] = ad;
}

__global__ void k_alpha(const int* __restrict__ ei, const float* __restrict__ attr,
                        const float* __restrict__ asrc, const float* __restrict__ adst,
                        const float* __restrict__ aeloop, const float* __restrict__ wev,
                        float* alpha, unsigned* amax){
  int e = blockIdx.x * blockDim.x + threadIdx.x;
  if (e >= ETOT) return;
  int s, dn; float ae;
  if (e < EE){
    s = ei[e]; dn = ei[EE + e];
    ae = 0.f;
    for (int d = 0; d < 8; ++d) ae += attr[(size_t)e * 8 + d] * wev[d];
  } else { s = dn = e - EE; ae = aeloop[s]; }
  float al = asrc[s] + adst[dn] + ae;
  al = al > 0.f ? al : 0.2f * al;          // leaky_relu(0.2)
  alpha[e] = al;
  atomicMax(&amax[dn], fenc(al));
}

__global__ void k_expden(const int* __restrict__ ei, float* alpha,
                         const unsigned* __restrict__ amax, float* den){
  int e = blockIdx.x * blockDim.x + threadIdx.x;
  if (e >= ETOT) return;
  int dn = (e < EE) ? ei[EE + e] : e - EE;
  float ex = __expf(alpha[e] - fdec(amax[dn]));
  alpha[e] = ex;
  atomicAdd(&den[dn], ex);
}

__global__ void k_scatter(const int* __restrict__ ei, const float* __restrict__ alpha,
                          const float* __restrict__ den, const float* __restrict__ xs,
                          float* out, int C){
  long long idx = (long long)blockIdx.x * blockDim.x + threadIdx.x;
  if (idx >= (long long)ETOT * C) return;
  int e = (int)(idx / C), c = (int)(idx % C);
  int s, dn;
  if (e < EE){ s = ei[e]; dn = ei[EE + e]; } else { s = dn = e - EE; }
  __builtin_prefetch(&xs[(size_t)s * C], 0, 1);   // global_prefetch_b8
  float w = alpha[e] / den[dn];
  atomicAdd(&out[(size_t)dn * C + c], w * xs[(size_t)s * C + c]);
}

// -------------------- batchnorm (training stats) --------------------
__global__ void k_bn_stats(const float* __restrict__ x, float* sums, int Nr, int C){
  int c = threadIdx.x % C;
  int rpb = blockDim.x / C;
  int r0 = blockIdx.x * rpb + threadIdx.x / C;
  int stride = gridDim.x * rpb;
  float s = 0.f, q = 0.f;
  for (int r = r0; r < Nr; r += stride){
    float v = x[(size_t)r * C + c];
    s += v; q += v * v;
  }
  atomicAdd(&sums[c], s);
  atomicAdd(&sums[C + c], q);
}

__global__ void k_bn_apply(float* x, const float* __restrict__ sums,
                           const float* __restrict__ g, const float* __restrict__ b,
                           int Nr, int C, int relu){
  long long idx = (long long)blockIdx.x * blockDim.x + threadIdx.x;
  if (idx >= (long long)Nr * C) return;
  int c = (int)(idx % C);
  float m = sums[c] / (float)Nr;
  float var = sums[C + c] / (float)Nr - m * m;
  float v = (x[idx] - m) * rsqrtf(var + 1e-5f) * g[c] + b[c];
  if (relu) v = fmaxf(v, 0.f);
  x[idx] = v;
}

// -------------------- conv1d k=3 s=3, 16->16 channels --------------------
// mode 1: input is node-major features [ROWS*250, 16]; mode 0: [ROWS,16,Lin]
__global__ void k_conv(const float* __restrict__ in, const float* __restrict__ w,
                       const float* __restrict__ bias, float* out,
                       int Lin, int Lout, int mode, const float* __restrict__ nodeFeat){
  long long idx = (long long)blockIdx.x * blockDim.x + threadIdx.x;
  long long total = (long long)ROWS * 16 * Lout;
  if (idx >= total) return;
  int l = (int)(idx % Lout);
  int o = (int)((idx / Lout) % 16);
  int b = (int)(idx / ((long long)Lout * 16));
  float s = bias[o];
  for (int i = 0; i < 16; ++i)
    for (int kk = 0; kk < 3; ++kk){
      int t = 3 * l + kk;
      float x = mode ? nodeFeat[((size_t)b * TT + t) * 16 + i]
                     : in[((size_t)b * 16 + i) * Lin + t];
      s += x * w[(o * 16 + i) * 3 + kk];
    }
  out[((size_t)b * 16 + o) * Lout + l] = s;
}

// -------------------- encoder projection + positional encoding --------------------
__global__ void k_encpe(const float* __restrict__ c3, const float* __restrict__ ew,
                        const float* __restrict__ eb, float* h0){
  long long idx = (long long)blockIdx.x * blockDim.x + threadIdx.x;
  if (idx >= (long long)SEQ * DM) return;
  int d = (int)(idx & 63);
  int row = (int)(idx >> 6);
  int b = row / 9, l = row % 9;
  float s = eb[d];
  for (int i = 0; i < 16; ++i)
    s += c3[((size_t)b * 16 + i) * 9 + l] * ew[i * DM + d];
  // reference quirk: pe indexed by batch-row dim
  int i2 = d & ~1;
  float div = __expf((float)i2 * (-9.210340371976184f / (float)DM));
  s += (d & 1) ? __cosf((float)b * div) : __sinf((float)b * div);
  h0[idx] = s;
}

// -------------------- generic WMMA GEMM: one wave per 16x16 tile --------------------
// C[m0+..,n0+..] = act(A @ B + bias [+ resid]); M,N mult of 16, K mult of 32.
// A tile staged to LDS via CDNA5 async global->LDS b128 copies.
__global__ __launch_bounds__(32) void k_gemm16(
    const float* __restrict__ A, int lda, const float* __restrict__ B, int ldb,
    const float* __restrict__ bias, const float* __restrict__ resid,
    float* __restrict__ C, int ldc, int K, int relu){
  __shared__ float Asm[16][32];
  unsigned lds_base = (unsigned)(size_t)&Asm[0][0];
  int lane = threadIdx.x & 31;
  int hf = lane >> 4, fr = lane & 15;
  int m0 = blockIdx.x * 16, n0 = blockIdx.y * 16;
  v8f acc;
  #pragma unroll
  for (int i = 0; i < 8; ++i) acc[i] = 0.f;
  for (int k0 = 0; k0 < K; k0 += 32){
    #pragma unroll
    for (int i = 0; i < 4; ++i){              // 16x32 f32 tile = 4 async b128 wave-issues
      int f = i * 128 + lane * 4;
      int r = f >> 5, cc = f & 31;
      async_b128(lds_base + (unsigned)f * 4u, &A[(size_t)(m0 + r) * lda + k0 + cc]);
    }
    async_wait0();
    __syncthreads();
    v16h af, bf;
    #pragma unroll
    for (int j = 0; j < 16; ++j){
      int kk = kmap16(j, hf);
      af[j] = (_Float16)Asm[fr][kk];
      bf[j] = (_Float16)B[(size_t)(k0 + kk) * ldb + n0 + fr];  // coalesced across lanes
    }
    acc = wmma_f16(af, bf, acc);
    __syncthreads();
  }
  int col = n0 + fr;
  #pragma unroll
  for (int i = 0; i < 8; ++i){
    int row = m0 + i + (hf ? 8 : 0);
    float v = acc[i] + bias[col];
    if (resid) v += resid[(size_t)row * ldc + col];
    if (relu) v = fmaxf(v, 0.f);
    C[(size_t)row * ldc + col] = v;
  }
}

// -------------------- per-sequence attention (L=9, D=64, 1 head) --------------------
__global__ __launch_bounds__(128) void k_attn(const float* __restrict__ qkv, float* o){
  int b = blockIdx.x, t = threadIdx.x;
  __shared__ float q[9][DM], k[9][DM], v[9][DM], att[9][9];
  for (int idx = t; idx < 9 * DM; idx += 128){
    int l = idx >> 6, d = idx & 63;
    const float* row = qkv + (size_t)(b * 9 + l) * (3 * DM);
    q[l][d] = row[d]; k[l][d] = row[DM + d]; v[l][d] = row[2 * DM + d];
  }
  __syncthreads();
  if (t < 81){
    int l = t / 9, m = t % 9;
    float s = 0.f;
    for (int d = 0; d < DM; ++d) s += q[l][d] * k[m][d];
    att[l][m] = s * 0.125f;                  // 1/sqrt(64)
  }
  __syncthreads();
  if (t < 9){
    float mx = -1e30f;
    for (int m = 0; m < 9; ++m) mx = fmaxf(mx, att[t][m]);
    float e[9], sm = 0.f;
    for (int m = 0; m < 9; ++m){ e[m] = __expf(att[t][m] - mx); sm += e[m]; }
    for (int m = 0; m < 9; ++m) att[t][m] = e[m] / sm;
  }
  __syncthreads();
  if (t < DM){
    for (int l = 0; l < 9; ++l){
      float s = 0.f;
      for (int m = 0; m < 9; ++m) s += att[l][m] * v[m][t];
      o[(size_t)(b * 9 + l) * DM + t] = s;
    }
  }
}

// -------------------- layernorm over 64, in place --------------------
__global__ __launch_bounds__(64) void k_ln(float* x, const float* __restrict__ g,
                                           const float* __restrict__ b){
  int r = blockIdx.x, t = threadIdx.x;
  __shared__ float sm[64];
  float v = x[(size_t)r * DM + t];
  sm[t] = v; __syncthreads();
  for (int s = 32; s; s >>= 1){ if (t < s) sm[t] += sm[t + s]; __syncthreads(); }
  float mean = sm[0] * (1.f / 64.f); __syncthreads();
  float d = v - mean;
  sm[t] = d * d; __syncthreads();
  for (int s = 32; s; s >>= 1){ if (t < s) sm[t] += sm[t + s]; __syncthreads(); }
  float var = sm[0] * (1.f / 64.f);
  x[(size_t)r * DM + t] = d * rsqrtf(var + 1e-5f) * g[t] + b[t];
}

// -------------------- fused FFN: Out = Hin + relu(Hin@W1+B1)@W2 + B2 --------------------
// 16-row tile per block; 16x2048 f16 intermediate lives entirely in LDS (64KB of 320KB WGP LDS).
// Input tile staged with one async b128 wave-issue per wave (tile is contiguous: lda == DM).
__global__ __launch_bounds__(256) void k_ffn(const float* __restrict__ Hin,
                                             const float* __restrict__ W1, const float* __restrict__ B1,
                                             const float* __restrict__ W2, const float* __restrict__ B2,
                                             float* __restrict__ Out){
  __shared__ float ht[16][DM];
  __shared__ _Float16 mid[16][FFD];
  int tid = threadIdx.x, lane = tid & 31, wave = tid >> 5;
  int hf = lane >> 4, fr = lane & 15;
  int m0 = blockIdx.x * 16;
  {
    int f = tid * 4;                           // 1024 floats total, 4 per thread
    unsigned htb = (unsigned)(size_t)&ht[0][0];
    async_b128(htb + (unsigned)f * 4u, &Hin[(size_t)m0 * DM + f]);
    async_wait0();
  }
  __syncthreads();
  // phase 1: mid = relu(ht @ W1 + B1), 8 waves x 16 N-tiles each
  for (int nt = wave; nt < FFD / 16; nt += 8){
    int n0 = nt * 16;
    v8f acc;
    #pragma unroll
    for (int i = 0; i < 8; ++i) acc[i] = 0.f;
    #pragma unroll
    for (int k0 = 0; k0 < DM; k0 += 32){
      v16h af, bf;
      #pragma unroll
      for (int j = 0; j < 16; ++j){
        int kk = k0 + kmap16(j, hf);
        af[j] = (_Float16)ht[fr][kk];
        bf[j] = (_Float16)W1[(size_t)kk * FFD + n0 + fr];
      }
      acc = wmma_f16(af, bf, acc);
    }
    #pragma unroll
    for (int i = 0; i < 8; ++i){
      int r = i + (hf ? 8 : 0);
      float v = fmaxf(acc[i] + B1[n0 + fr], 0.f);
      mid[r][n0 + fr] = (_Float16)v;
    }
  }
  __syncthreads();
  // phase 2: Out tile = mid @ W2 + B2 + residual, waves 0..3 (one 16-col tile each)
  if (wave < 4){
    int n0 = wave * 16;
    v8f acc;
    #pragma unroll
    for (int i = 0; i < 8; ++i) acc[i] = 0.f;
    for (int k0 = 0; k0 < FFD; k0 += 32){
      v16h af, bf;
      #pragma unroll
      for (int j = 0; j < 16; ++j){
        int kk = k0 + kmap16(j, hf);
        af[j] = mid[fr][kk];
        bf[j] = (_Float16)W2[(size_t)kk * DM + n0 + fr];
      }
      acc = wmma_f16(af, bf, acc);
    }
    #pragma unroll
    for (int i = 0; i < 8; ++i){
      int row = m0 + i + (hf ? 8 : 0);
      int col = n0 + fr;
      Out[(size_t)row * DM + col] = acc[i] + B2[col] + Hin[(size_t)row * DM + col];
    }
  }
}

// -------------------- mean pool per graph (62 rows) + fc + sigmoid --------------------
__global__ __launch_bounds__(64) void k_pool_fc(const float* __restrict__ h,
                                                const float* __restrict__ fcw,
                                                const float* __restrict__ fcb, float* out){
  int t = threadIdx.x;
  if (t >= 64) return;
  int g = t >> 1, j = t & 1;
  float acc = fcb[j];
  for (int c = 0; c < 32; ++c){
    float s = 0.f;
    for (int r = 0; r < 62; ++r) s += h[(size_t)(g * 62 + r) * 32 + c];
    acc += (s * (1.f / 62.f)) * fcw[c * 2 + j];
  }
  out[g * 2 + j] = 1.f / (1.f + __expf(-acc));
}

// ==================== host orchestration ====================
extern "C" void kernel_launch(void* const* d_in, const int* in_sizes, int n_in,
                              void* d_out, int out_size, void* d_ws, size_t ws_size,
                              hipStream_t stream){
  (void)in_sizes; (void)n_in; (void)out_size; (void)ws_size;
  const float* X    = (const float*)d_in[0];
  const int*   ei   = (const int*)d_in[1];
  const float* attr = (const float*)d_in[2];
  // d_in[3] = batch (contiguous 62-per-graph grouping, used implicitly in k_pool_fc)
  auto P = [&](int i){ return (const float*)d_in[i]; };

  float* ws = (float*)d_ws;
  size_t off = 0;
  auto alloc = [&](size_t n){ float* p = ws + off; off += (n + 255) & ~(size_t)255; return p; };
  float* nodeA  = alloc((size_t)NN * 32);
  float* nodeB  = alloc((size_t)NN * 32);
  float* nodeC  = alloc((size_t)NN * 32);
  float* loops  = alloc((size_t)NN * 8);
  float* cnt    = alloc(NN);
  float* asrc   = alloc(NN);
  float* adst   = alloc(NN);
  float* amaxb  = alloc(NN);
  float* den    = alloc(NN);
  float* aeloop = alloc(NN);
  float* alpha  = alloc(ETOT);
  float* wev    = alloc(256);
  float* bns    = alloc(256);
  float* h0     = alloc((size_t)SEQ * DM);
  float* qkv    = alloc((size_t)SEQ * 3 * DM);
  float* ao     = alloc((size_t)SEQ * DM);
  float* h1     = alloc((size_t)SEQ * DM);
  float* h2     = alloc((size_t)SEQ * DM);
  float* c1     = alloc((size_t)ROWS * 16 * 83);
  float* c2     = alloc((size_t)ROWS * 16 * 27);
  float* c3     = alloc((size_t)ROWS * 16 * 9);
  float* decb   = alloc((size_t)ROWS * 32);

  auto gb = [](long long n){ return (unsigned)((n + 255) / 256); };

  // self-loop edge-attr mean (once)
  k_zero<<<gb((long long)NN * 8), 256, 0, stream>>>(loops, NN * 8);
  k_zero<<<gb(NN), 256, 0, stream>>>(cnt, NN);
  k_loop_attr<<<gb((long long)EE * 8), 256, 0, stream>>>(ei, attr, loops, cnt);

  auto gat = [&](const float* hin, int Cin, int C, int pbase,
                 const float* bng, const float* bnb, float* xs, float* out){
    k_edge_vec<<<1, 32, 0, stream>>>(P(pbase + 3), P(pbase + 4), wev, C);
    k_aeloop<<<gb(NN), 256, 0, stream>>>(loops, cnt, wev, aeloop);
    k_xs<<<gb(NN), 256, 0, stream>>>(hin, P(pbase + 0), P(pbase + 1), P(pbase + 2),
                                     xs, asrc, adst, Cin, C);
    k_zero<<<gb(NN), 256, 0, stream>>>(amaxb, NN);    // 0u == encoded minimum
    k_zero<<<gb(NN), 256, 0, stream>>>(den, NN);
    k_zero<<<gb((long long)NN * C), 256, 0, stream>>>(out, NN * C);
    k_alpha<<<gb(ETOT), 256, 0, stream>>>(ei, attr, asrc, adst, aeloop, wev,
                                          alpha, (unsigned*)amaxb);
    k_expden<<<gb(ETOT), 256, 0, stream>>>(ei, alpha, (const unsigned*)amaxb, den);
    k_scatter<<<gb((long long)ETOT * C), 256, 0, stream>>>(ei, alpha, den, xs, out, C);
    k_zero<<<1, 64, 0, stream>>>(bns, 64);
    k_bn_stats<<<512, 256, 0, stream>>>(out, bns, NN, C);
    k_bn_apply<<<gb((long long)NN * C), 256, 0, stream>>>(out, bns, bng, bnb, NN, C, 1);
  };
  gat(X,     5, 32, 4,  P(22), P(23), nodeB, nodeA);   // gat1 + bn1 + relu
  gat(nodeA, 32, 16, 10, P(24), P(25), nodeB, nodeC);  // gat2 + bn2 + relu
  gat(nodeC, 16, 16, 16, P(26), P(27), nodeB, nodeA);  // gat3 + bn3 + relu

  // conv stack (stride-3, VALID): 250 -> 83 -> 27 -> 9
  k_conv<<<gb((long long)ROWS * 16 * 83), 256, 0, stream>>>(nullptr, P(28), P(29), c1, 250, 83, 1, nodeA);
  k_conv<<<gb((long long)ROWS * 16 * 27), 256, 0, stream>>>(c1, P(30), P(31), c2, 83, 27, 0, nullptr);
  k_conv<<<gb((long long)ROWS * 16 * 9),  256, 0, stream>>>(c2, P(32), P(33), c3, 27, 9, 0, nullptr);

  // transformer tail (WMMA territory)
  k_encpe<<<gb((long long)SEQ * DM), 256, 0, stream>>>(c3, P(34), P(35), h0);
  k_gemm16<<<dim3(SEQ / 16, 192 / 16), 32, 0, stream>>>(h0, DM, P(36), 3 * DM, P(37),
                                                        nullptr, qkv, 3 * DM, DM, 0);
  k_attn<<<ROWS, 128, 0, stream>>>(qkv, ao);
  k_gemm16<<<dim3(SEQ / 16, DM / 16), 32, 0, stream>>>(ao, DM, P(38), DM, P(39),
                                                       h0, h1, DM, DM, 0);   // + residual
  k_ln<<<SEQ, 64, 0, stream>>>(h1, P(40), P(41));
  k_ffn<<<SEQ / 16, 256, 0, stream>>>(h1, P(42), P(43), P(44), P(45), h2);
  k_ln<<<SEQ, 64, 0, stream>>>(h2, P(46), P(47));

  // decoder on last token (row stride 9*64, offset +8*64)
  k_gemm16<<<dim3(ROWS / 16, 32 / 16), 32, 0, stream>>>(h2 + 8 * DM, 9 * DM, P(48), 32,
                                                        P(49), nullptr, decb, 32, DM, 0);
  k_zero<<<1, 64, 0, stream>>>(bns, 64);
  k_bn_stats<<<512, 256, 0, stream>>>(decb, bns, ROWS, 32);
  k_bn_apply<<<gb((long long)ROWS * 32), 256, 0, stream>>>(decb, bns, P(50), P(51), ROWS, 32, 0);

  k_pool_fc<<<1, 64, 0, stream>>>(decb, P(52), P(53), (float*)d_out);
}
